// RNNModel_35751307771982
// MI455X (gfx1250) — matrix operand order repaired and checked by reading, not compiled, not run on hardware
//
#include <hip/hip_runtime.h>
#include <hip/hip_bf16.h>

// ---------------------------------------------------------------------------
// Sizes (from reference)
// ---------------------------------------------------------------------------
#define SEQ    128
#define BS     32
#define SB     (SEQ * BS)      // 4096 rows
#define NINP   1024
#define NHID   1024
#define NG     (4 * NHID)      // 4096 gate columns
#define NTOK   50257
#define NTOKP  50272           // padded to multiple of 32

typedef unsigned short u16;    // raw bf16 storage
typedef unsigned int   u32;

typedef __attribute__((ext_vector_type(16))) __bf16 v16bf;
typedef __attribute__((ext_vector_type(8)))  float  v8f;

struct alignas(16) U4 { u32 x, y, z, w; };
union BfTile { v16bf v; U4 q[2]; };

// ---------------------------------------------------------------------------
// Helpers
// ---------------------------------------------------------------------------
__device__ __forceinline__ u16 f2bf(float f) {
    u32 u = __float_as_uint(f);
    u32 r = (u + 0x7FFFu + ((u >> 16) & 1u)) >> 16;   // round-to-nearest-even
    return (u16)r;
}

__device__ __forceinline__ float sigf(float x) {
    return 1.0f / (1.0f + __expf(-x));
}

__device__ __forceinline__ v8f wmma_bf16(v16bf a, v16bf b, v8f c) {
    return __builtin_amdgcn_wmma_f32_16x16x32_bf16(
        /*neg_a=*/false, a, /*neg_b=*/false, b,
        /*c_mod=*/(short)0, c, /*reuse_a=*/false, /*reuse_b=*/false);
}

// A tile (16x32 bf16 MxK): lane -> M = lane&15; K-half by lane>=16.
// pa is the per-lane base: &A[(m0 + lane&15)*lda + ((lane>>4)<<3)].
__device__ __forceinline__ void load_A(BfTile& t, const u16* pa) {
    t.q[0] = *(const U4*)(pa);
    t.q[1] = *(const U4*)(pa + 16);
}
// B tile (32x16 bf16 KxN): lane -> N = lane&15; K-half by lane>=16 (table order).
// pb is the per-lane base: &Bt[(k0 + ((lane>>4)<<3))*ldb + n0 + (lane&15)].
__device__ __forceinline__ void load_B(BfTile& t, const u16* pb, size_t ldb) {
    u16* e = (u16*)&t;
#pragma unroll
    for (int s = 0; s < 8; ++s) {
        e[s]     = pb[(size_t)s * ldb];
        e[s + 8] = pb[(size_t)(16 + s) * ldb];
    }
}

// ---------------------------------------------------------------------------
// Data-prep kernels
// ---------------------------------------------------------------------------

// out[C, Rpad] (bf16) = transpose(in[R, C] fp32), zero-pad rows R..Rpad-1.
__global__ __launch_bounds__(256) void transpose_cvt(
    const float* __restrict__ in, u16* __restrict__ out, int R, int C, int Rpad)
{
    size_t idx = (size_t)blockIdx.x * 256 + threadIdx.x;
    size_t total = (size_t)C * Rpad;
    if (idx >= total) return;
    int k = (int)(idx / Rpad);
    int n = (int)(idx % Rpad);
    out[idx] = (n < R) ? f2bf(in[(size_t)n * C + k]) : (u16)0;
}

// emb[row, :] = bf16(emb_w[tokens[row], :])
__global__ __launch_bounds__(256) void embed_gather(
    const int* __restrict__ tokens, const float* __restrict__ emb_w,
    u16* __restrict__ out)
{
    int row = blockIdx.x;
    int tok = tokens[row];
    const float* src = emb_w + (size_t)tok * NINP;
    u16* dst = out + (size_t)row * NINP;
    for (int k = threadIdx.x; k < NINP; k += 256) dst[k] = f2bf(src[k]);
}

__global__ __launch_bounds__(256) void cvt_f32_bf16(
    const float* __restrict__ in, u16* __restrict__ out, int n)
{
    int i = blockIdx.x * 256 + threadIdx.x;
    if (i < n) out[i] = f2bf(in[i]);
}

__global__ __launch_bounds__(256) void copy_f32(
    const float* __restrict__ in, float* __restrict__ out, int n)
{
    int i = blockIdx.x * 256 + threadIdx.x;
    if (i < n) out[i] = in[i];
}

__global__ __launch_bounds__(256) void zero_u32(u32* __restrict__ p, int n)
{
    int i = blockIdx.x * 256 + threadIdx.x;
    if (i < n) p[i] = 0u;
}

// ---------------------------------------------------------------------------
// GEMM: C[M, ldc] = A[M, K](bf16) @ Bt[K, Npad](bf16) + bias[n]
// Block = 256 threads = 8 waves. Each wave computes a 32x32 output slab
// (2 M-tiles x 2 N-tiles -> 4 WMMAs per K-chunk from 4 A b128 + 32 B u16 loads).
// Software-pipelined: next-chunk loads are issued before current WMMAs.
// Requires ldb (Npad) multiple of 32, K multiple of 32, M multiple of 32.
// ---------------------------------------------------------------------------
__global__ __launch_bounds__(256) void gemm_bf16_bias(
    const u16* __restrict__ A, int lda,
    const u16* __restrict__ Bt, int ldb,     // ldb == Npad
    const float* __restrict__ bias,
    float* __restrict__ C, int ldc,
    int Ncols, int K)
{
    int lane = threadIdx.x & 31;
    int wid  = threadIdx.x >> 5;
    int n0 = (blockIdx.x * 8 + wid) * 32;    // 2 N tiles
    if (n0 >= ldb) return;
    int m0 = blockIdx.y * 32;                // 2 M tiles

    // per-lane base pointers (constant offsets inside the loop)
    const u16* pa0 = A + (size_t)(m0 + (lane & 15)) * lda + ((lane >> 4) << 3);
    const u16* pa1 = pa0 + (size_t)16 * lda;
    const u16* pb  = Bt + (size_t)((lane >> 4) << 3) * ldb + n0 + (lane & 15);
    const size_t bstep = (size_t)32 * ldb;   // K-chunk stride for B

    v8f acc00 = {}, acc01 = {}, acc10 = {}, acc11 = {};

    BfTile a0, a1, b0, b1;
    load_A(a0, pa0);
    load_A(a1, pa1);
    load_B(b0, pb, ldb);
    load_B(b1, pb + 16, ldb);

    for (int k0 = 32; k0 < K; k0 += 32) {
        BfTile na0, na1, nb0, nb1;
        const u16* qb = pb + (size_t)(k0 >> 5) * bstep;
        load_A(na0, pa0 + k0);
        load_A(na1, pa1 + k0);
        load_B(nb0, qb, ldb);
        load_B(nb1, qb + 16, ldb);

        acc00 = wmma_bf16(a0.v, b0.v, acc00);
        acc01 = wmma_bf16(a0.v, b1.v, acc01);
        acc10 = wmma_bf16(a1.v, b0.v, acc10);
        acc11 = wmma_bf16(a1.v, b1.v, acc11);

        a0 = na0; a1 = na1; b0 = nb0; b1 = nb1;
    }
    acc00 = wmma_bf16(a0.v, b0.v, acc00);
    acc01 = wmma_bf16(a0.v, b1.v, acc01);
    acc10 = wmma_bf16(a1.v, b0.v, acc10);
    acc11 = wmma_bf16(a1.v, b1.v, acc11);

    // store: lane -> column n0 + lane&15 (and +16); 8 rows per lane per tile
    int colA = n0 + (lane & 15);
    int colB = colA + 16;
    int roff = (lane >> 4) << 3;             // 0 or 8
    float* r0 = C + (size_t)(m0 + roff) * ldc;
    float* r1 = C + (size_t)(m0 + 16 + roff) * ldc;
    if (colA < Ncols) {
        float bs = bias ? bias[colA] : 0.0f;
#pragma unroll
        for (int v = 0; v < 8; ++v) {
            r0[(size_t)v * ldc + colA] = acc00[v] + bs;
            r1[(size_t)v * ldc + colA] = acc10[v] + bs;
        }
    }
    if (colB < Ncols) {
        float bs = bias ? bias[colB] : 0.0f;
#pragma unroll
        for (int v = 0; v < 8; ++v) {
            r0[(size_t)v * ldc + colB] = acc01[v] + bs;
            r1[(size_t)v * ldc + colB] = acc11[v] + bs;
        }
    }
}

// ---------------------------------------------------------------------------
// One LSTM time step: gates = Xg_t + h_in @ WhT; fused cell update.
// grid.x = NHID/16 = 64 blocks; block = 64 threads = 2 waves (wave -> 16 rows).
// Each wave owns gate columns j0..j0+15 for all four gates -> local update.
// Software-pipelined like the GEMM.
// ---------------------------------------------------------------------------
__global__ __launch_bounds__(64) void lstm_step(
    const float* __restrict__ Xg_t,   // [BS x NG]
    const u16*   __restrict__ WhT,    // [NHID x NG] bf16 (pre-transposed)
    const u16*   __restrict__ h_in,   // [BS x NHID] bf16
    const float* __restrict__ c_in,   // [BS x NHID]
    u16*         __restrict__ h_out,  // [BS x NHID] bf16 (hs[t] slice)
    float*       __restrict__ h_out_f,// [BS x NHID]
    float*       __restrict__ c_out)  // [BS x NHID]
{
    int lane = threadIdx.x & 31;
    int wid  = threadIdx.x >> 5;      // 0..1
    int m0 = wid * 16;                // batch-row tile
    int j0 = blockIdx.x * 16;         // hidden-unit tile

    const u16* pa = h_in + (size_t)(m0 + (lane & 15)) * NHID + ((lane >> 4) << 3);
    const u16* pb = WhT + (size_t)((lane >> 4) << 3) * NG + j0 + (lane & 15);
    const size_t bstep = (size_t)32 * NG;

    v8f acc[4] = {{}, {}, {}, {}};

    BfTile a, b[4];
    load_A(a, pa);
#pragma unroll
    for (int g = 0; g < 4; ++g) load_B(b[g], pb + g * NHID, NG);

    for (int k0 = 32; k0 < NHID; k0 += 32) {
        BfTile na, nb[4];
        const u16* qb = pb + (size_t)(k0 >> 5) * bstep;
        load_A(na, pa + k0);
#pragma unroll
        for (int g = 0; g < 4; ++g) load_B(nb[g], qb + g * NHID, NG);

#pragma unroll
        for (int g = 0; g < 4; ++g) acc[g] = wmma_bf16(a.v, b[g].v, acc[g]);

        a = na;
#pragma unroll
        for (int g = 0; g < 4; ++g) b[g] = nb[g];
    }
#pragma unroll
    for (int g = 0; g < 4; ++g) acc[g] = wmma_bf16(a.v, b[g].v, acc[g]);

    int j  = j0 + (lane & 15);
    int b0r = m0 + ((lane >> 4) << 3);
#pragma unroll
    for (int v = 0; v < 8; ++v) {
        int brow = b0r + v;
        const float* xg = Xg_t + (size_t)brow * NG;
        float gi = acc[0][v] + xg[j];
        float gf = acc[1][v] + xg[NHID + j];
        float gg = acc[2][v] + xg[2 * NHID + j];
        float go = acc[3][v] + xg[3 * NHID + j];
        float cc = sigf(gf) * c_in[(size_t)brow * NHID + j] + sigf(gi) * tanhf(gg);
        float hh = sigf(go) * tanhf(cc);
        c_out[(size_t)brow * NHID + j]   = cc;
        h_out_f[(size_t)brow * NHID + j] = hh;
        h_out[(size_t)brow * NHID + j]   = f2bf(hh);
    }
}

// ---------------------------------------------------------------------------
// log-softmax: per-row max + log-sum-exp, then subtract.
// ---------------------------------------------------------------------------
__global__ __launch_bounds__(256) void rowstats(
    const float* __restrict__ logits, float* __restrict__ stats, int N)
{
    __shared__ float red[256];
    int row = blockIdx.x;
    int tid = threadIdx.x;
    const float* x = logits + (size_t)row * N;

    float m = -3.4e38f;
    for (int i = tid; i < N; i += 256) m = fmaxf(m, x[i]);
    red[tid] = m; __syncthreads();
    for (int s = 128; s > 0; s >>= 1) {
        if (tid < s) red[tid] = fmaxf(red[tid], red[tid + s]);
        __syncthreads();
    }
    float rowmax = red[0]; __syncthreads();

    float sum = 0.0f;
    for (int i = tid; i < N; i += 256) sum += __expf(x[i] - rowmax);
    red[tid] = sum; __syncthreads();
    for (int s = 128; s > 0; s >>= 1) {
        if (tid < s) red[tid] += red[tid + s];
        __syncthreads();
    }
    if (tid == 0) stats[row] = rowmax + __logf(red[0]);
}

__global__ __launch_bounds__(256) void lsm_apply(
    float* __restrict__ logits, const float* __restrict__ stats, int N)
{
    int row = blockIdx.x;
    float s = stats[row];
    float* x = logits + (size_t)row * N;
    for (int i = threadIdx.x; i < N; i += 256) x[i] -= s;
}

// ---------------------------------------------------------------------------
// Host-side orchestration
// ---------------------------------------------------------------------------
static inline size_t alignup(size_t x) { return (x + 255) & ~(size_t)255; }

extern "C" void kernel_launch(void* const* d_in, const int* in_sizes, int n_in,
                              void* d_out, int out_size, void* d_ws, size_t ws_size,
                              hipStream_t stream) {
    (void)in_sizes; (void)n_in; (void)out_size; (void)ws_size;

    const int*   tokens = (const int*)  d_in[0];
    const float* h0     = (const float*)d_in[1];
    const float* c0     = (const float*)d_in[2];
    const float* emb_w  = (const float*)d_in[3];
    const float* Wx1    = (const float*)d_in[4];
    const float* bx1    = (const float*)d_in[5];
    const float* Wh1    = (const float*)d_in[6];
    const float* Wx2    = (const float*)d_in[7];
    const float* bx2    = (const float*)d_in[8];
    const float* Wh2    = (const float*)d_in[9];
    const float* dec_w  = (const float*)d_in[10];
    const float* dec_b  = (const float*)d_in[11];
    float* out = (float*)d_out;

    // ---- carve workspace (needs ~221 MB) ----
    char* ws = (char*)d_ws;
    size_t off = 0;
    auto take = [&](size_t bytes) { char* p = ws + off; off = alignup(off + bytes); return p; };

    u16*   embA  = (u16*)  take((size_t)SB * NINP * 2);      // bf16 embeddings
    u16*   hs1   = (u16*)  take((size_t)SB * NHID * 2);      // layer-1 hidden seq
    u16*   hs2   = (u16*)  take((size_t)SB * NHID * 2);      // layer-2 hidden seq
    u16*   WxT1  = (u16*)  take((size_t)NINP * NG * 2);
    u16*   WhT1  = (u16*)  take((size_t)NHID * NG * 2);
    u16*   WxT2  = (u16*)  take((size_t)NHID * NG * 2);
    u16*   WhT2  = (u16*)  take((size_t)NHID * NG * 2);
    u16*   decT  = (u16*)  take((size_t)NHID * NTOKP * 2);   // dec_w^T, padded
    float* Xg    = (float*)take((size_t)SB * NG * 4);        // shared by both layers
    u16*   h0l1  = (u16*)  take((size_t)BS * NHID * 2);
    u16*   h0l2  = (u16*)  take((size_t)BS * NHID * 2);
    float* c1b0  = (float*)take((size_t)BS * NHID * 4);
    float* c1b1  = (float*)take((size_t)BS * NHID * 4);
    float* c2b0  = (float*)take((size_t)BS * NHID * 4);
    float* c2b1  = (float*)take((size_t)BS * NHID * 4);
    float* h1f   = (float*)take((size_t)BS * NHID * 4);
    float* h2f   = (float*)take((size_t)BS * NHID * 4);
    float* stats = (float*)take((size_t)SB * 4);
    float* c1buf[2] = {c1b0, c1b1};
    float* c2buf[2] = {c2b0, c2b1};

    const int HC = BS * NHID;  // 32768

    // ---- weight conversion / transposition (bf16, [K, Npad]) ----
    {
        int blk = (NINP * NG + 255) / 256;
        transpose_cvt<<<blk, 256, 0, stream>>>(Wx1, WxT1, NG, NINP, NG);
        transpose_cvt<<<blk, 256, 0, stream>>>(Wh1, WhT1, NG, NHID, NG);
        transpose_cvt<<<blk, 256, 0, stream>>>(Wx2, WxT2, NG, NHID, NG);
        transpose_cvt<<<blk, 256, 0, stream>>>(Wh2, WhT2, NG, NHID, NG);
        size_t tot = (size_t)NHID * NTOKP;
        transpose_cvt<<<(int)((tot + 255) / 256), 256, 0, stream>>>(dec_w, decT, NTOK, NHID, NTOKP);
    }

    // ---- embedding gather ----
    embed_gather<<<SB, 256, 0, stream>>>(tokens, emb_w, embA);

    // ---- layer 1: Xg = emb @ Wx1^T + bx1 (time-parallel) ----
    gemm_bf16_bias<<<dim3(NG / 32 / 8, SB / 32), 256, 0, stream>>>(
        embA, NINP, WxT1, NG, bx1, Xg, NG, NG, NINP);

    // ---- layer-1 state init ----
    cvt_f32_bf16<<<(HC + 255) / 256, 256, 0, stream>>>(h0, h0l1, HC);
    copy_f32<<<(HC + 255) / 256, 256, 0, stream>>>(c0, c1buf[0], HC);

    // ---- layer-1 recurrence ----
    for (int t = 0; t < SEQ; ++t) {
        const u16* hin = (t == 0) ? h0l1 : (hs1 + (size_t)(t - 1) * HC);
        lstm_step<<<NHID / 16, 64, 0, stream>>>(
            Xg + (size_t)t * BS * NG, WhT1, hin, c1buf[t & 1],
            hs1 + (size_t)t * HC, h1f, c1buf[(t + 1) & 1]);
    }

    // ---- layer 2: Xg = hs1 @ Wx2^T + bx2 ----
    gemm_bf16_bias<<<dim3(NG / 32 / 8, SB / 32), 256, 0, stream>>>(
        hs1, NHID, WxT2, NG, bx2, Xg, NG, NG, NHID);

    // ---- layer-2 state init (zeros) ----
    zero_u32<<<(HC / 2 + 255) / 256, 256, 0, stream>>>((u32*)h0l2, HC / 2);
    zero_u32<<<(HC + 255) / 256, 256, 0, stream>>>((u32*)c2buf[0], HC);

    // ---- layer-2 recurrence ----
    for (int t = 0; t < SEQ; ++t) {
        const u16* hin = (t == 0) ? h0l2 : (hs2 + (size_t)(t - 1) * HC);
        lstm_step<<<NHID / 16, 64, 0, stream>>>(
            Xg + (size_t)t * BS * NG, WhT2, hin, c2buf[t & 1],
            hs2 + (size_t)t * HC, h2f, c2buf[(t + 1) & 1]);
    }

    // ---- decoder: logits = hs2 @ dec_w^T + dec_b (into d_out) ----
    gemm_bf16_bias<<<dim3(NTOKP / 32 / 8 + 1, SB / 32), 256, 0, stream>>>(
        hs2, NHID, decT, NTOKP, dec_b, out, NTOK, NTOK, NHID);

    // ---- log-softmax over vocab ----
    rowstats<<<SB, 256, 0, stream>>>(out, stats, NTOK);
    lsm_apply<<<SB, 256, 0, stream>>>(out, stats, NTOK);

    // ---- tail outputs: h2, c2 ----
    float* tail = out + (size_t)SB * NTOK;
    copy_f32<<<(HC + 255) / 256, 256, 0, stream>>>(h2f, tail, HC);
    copy_f32<<<(HC + 255) / 256, 256, 0, stream>>>(c2buf[SEQ & 1], tail + HC, HC);
}
